// MultiHeadAttention_87935160418700
// MI455X (gfx1250) — matrix operand order repaired and checked by reading, not compiled, not run on hardware
//
#include <hip/hip_runtime.h>
#include <hip/hip_bf16.h>

// ---------------------------------------------------------------------------
// MHA forward for MI455X (gfx1250, wave32, WMMA f32_16x16x32_f16,
// async global->LDS staging with double buffering)
// B=2, S=2048, D=2048, H=16, Dh=128
// ---------------------------------------------------------------------------

typedef _Float16 v16h __attribute__((ext_vector_type(16)));
typedef _Float16 v8h  __attribute__((ext_vector_type(8)));
typedef _Float16 v4h  __attribute__((ext_vector_type(4)));
typedef float    v8f  __attribute__((ext_vector_type(8)));

#define DM      2048
#define NHEADS  16
#define DH      128
#define BATCH   2
#define SEQ     2048
// log2(e) / sqrt(128): fold attention scale + exp2 conversion into Q
#define QSCALE  0.1275337855f

static __device__ __forceinline__ v16h cat16(v8h lo, v8h hi) {
  v16h r;
#pragma unroll
  for (int i = 0; i < 8; ++i) { r[i] = lo[i]; r[i + 8] = hi[i]; }
  return r;
}

static __device__ __forceinline__ v8f wmma16(v16h a, v16h b, v8f c) {
  // D = A(16x32 f16) * B(32x16 f16) + C(16x16 f32)
  return __builtin_amdgcn_wmma_f32_16x16x32_f16(false, a, false, b,
                                                (short)0, c, false, false);
}

// Async copy of 16 bytes global -> LDS (ASYNCcnt tracked, bypasses VGPRs).
static __device__ __forceinline__ void async_ld16(const _Float16* g,
                                                  _Float16* l) {
  unsigned laddr = (unsigned)(size_t)(void*)l;  // low 32 bits = LDS offset
  asm volatile("global_load_async_to_lds_b128 %0, %1, off"
               :
               : "v"(laddr), "v"((unsigned long long)(size_t)g)
               : "memory");
}
static __device__ __forceinline__ void wait_async0() {
  asm volatile("s_wait_asynccnt 0" ::: "memory");
}

// ---------------------------------------------------------------------------
// fp32 -> fp16 conversion (vectorized x4)
// ---------------------------------------------------------------------------
__global__ __launch_bounds__(256) void cvt_f32_f16(const float* __restrict__ in,
                                                   _Float16* __restrict__ out,
                                                   int n4) {
  int i = blockIdx.x * 256 + threadIdx.x;
  if (i < n4) {
    float4 f = ((const float4*)in)[i];
    v4h h;
    h[0] = (_Float16)f.x; h[1] = (_Float16)f.y;
    h[2] = (_Float16)f.z; h[3] = (_Float16)f.w;
    ((v4h*)out)[i] = h;
  }
}

// ---------------------------------------------------------------------------
// Tiled GEMM: C[M,N] = A[M,K] * W[N,K]^T   (f16 in, f32 accum)
// Block tile 128x128, k-step 32, 8 waves; each wave: 2x4 grid of 16x16 WMMA.
// Double-buffered LDS; next k-tile prefetched via async-to-LDS while current
// tile is consumed by WMMA. One barrier + one asynccnt-wait per k-step.
// MODE 0: store f32 to Cout.
// MODE 1: QKV epilogue -> scatter f16 into Q/K/V [B,H,S,Dh]; Q scaled.
// ---------------------------------------------------------------------------
template <int MODE>
__global__ __launch_bounds__(256) void gemm_wmma(
    const _Float16* __restrict__ A, const _Float16* __restrict__ W,
    float* __restrict__ Cout, _Float16* __restrict__ Qh,
    _Float16* __restrict__ Kh, _Float16* __restrict__ Vh, int M, int N,
    int K) {
  constexpr int BK = 32;
  constexpr int LDT = BK + 8;  // padded LDS row stride (halfs)
  __shared__ alignas(16) _Float16 As[2][128][LDT];
  __shared__ alignas(16) _Float16 Bs[2][128][LDT];

  const int tid = threadIdx.x;
  const int lane = tid & 31;
  const int wave = tid >> 5;
  const int wm = wave >> 1;   // 0..3 (M direction, 32 rows each)
  const int wn = wave & 1;    // 0..1 (N direction, 64 cols each)
  const int lm = lane & 15;
  const int hi = lane >> 4;
  const int blockM = blockIdx.y * 128;
  const int blockN = blockIdx.x * 128;

  // Staging assignment: thread covers rows r0 and r0+64 at k-chunk ko (8h).
  const int r0 = tid >> 2;
  const int ko = (tid & 3) << 3;
  const _Float16* Ag = A + (size_t)(blockM + r0) * K + ko;
  const _Float16* Bg = W + (size_t)(blockN + r0) * K + ko;
  const size_t rowSkip = (size_t)64 * K;

  auto stage = [&](int buf, int kb) {
    async_ld16(Ag + kb, &As[buf][r0][ko]);
    async_ld16(Ag + kb + rowSkip, &As[buf][r0 + 64][ko]);
    async_ld16(Bg + kb, &Bs[buf][r0][ko]);
    async_ld16(Bg + kb + rowSkip, &Bs[buf][r0 + 64][ko]);
  };

  stage(0, 0);

  v8f acc[2][4] = {};
  const int nk = K / BK;
  for (int it = 0; it < nk; ++it) {
    const int cur = it & 1;
    wait_async0();     // this wave's fills of buf `cur` are in LDS
    __syncthreads();   // everyone's fills done; prev buffer free
    if (it + 1 < nk) stage(cur ^ 1, (it + 1) * BK);  // prefetch next tile

    // A fragments: 16x32, row = lane&15, K split per ISA layout
    v16h af[2];
#pragma unroll
    for (int mt = 0; mt < 2; ++mt) {
      int r = wm * 32 + mt * 16 + lm;
      v8h lo = *(const v8h*)&As[cur][r][hi * 8];
      v8h h8 = *(const v8h*)&As[cur][r][16 + hi * 8];
      af[mt] = cat16(lo, h8);
    }
    // B fragments: 32x16 (W^T tile), K split per ISA layout (same as A)
    v16h bf[4];
#pragma unroll
    for (int nt = 0; nt < 4; ++nt) {
      int r = wn * 64 + nt * 16 + lm;
      v8h lo = *(const v8h*)&Bs[cur][r][hi * 8];
      v8h h8 = *(const v8h*)&Bs[cur][r][16 + hi * 8];
      bf[nt] = cat16(lo, h8);
    }
#pragma unroll
    for (int mt = 0; mt < 2; ++mt)
#pragma unroll
      for (int nt = 0; nt < 4; ++nt)
        acc[mt][nt] = wmma16(af[mt], bf[nt], acc[mt][nt]);
  }

  // Epilogue: C frag row = r + 8*hi, col = lane&15
#pragma unroll
  for (int mt = 0; mt < 2; ++mt) {
#pragma unroll
    for (int nt = 0; nt < 4; ++nt) {
      int nCol = blockN + wn * 64 + nt * 16 + lm;
#pragma unroll
      for (int r = 0; r < 8; ++r) {
        int m = blockM + wm * 32 + mt * 16 + 8 * hi + r;
        float v = acc[mt][nt][r];
        if (MODE == 0) {
          Cout[(size_t)m * N + nCol] = v;
        } else {
          // n in [0,3D): which = q/k/v, then head h and dh
          int which = nCol >> 11;         // / 2048
          int d = nCol & (DM - 1);
          int h = d >> 7;                 // / 128
          int dh = d & (DH - 1);
          int b = m >> 11;                // / SEQ
          int s = m & (SEQ - 1);
          size_t dst = (((size_t)b * NHEADS + h) * SEQ + s) * DH + dh;
          if (which == 0)
            Qh[dst] = (_Float16)(v * QSCALE);
          else if (which == 1)
            Kh[dst] = (_Float16)v;
          else
            Vh[dst] = (_Float16)v;
        }
      }
    }
  }
}

// ---------------------------------------------------------------------------
// Flash attention: one block = 64 queries of one (b,h); 4 waves, each one
// 16-query tile. Key blocks of 32. exp2-domain softmax (scale folded into Q).
// Q and K staged with async global->LDS; V transposed through registers.
// ---------------------------------------------------------------------------
__global__ __launch_bounds__(128) void flash_attn_wmma(
    const _Float16* __restrict__ Qh, const _Float16* __restrict__ Kh,
    const _Float16* __restrict__ Vh, _Float16* __restrict__ Oh) {
  constexpr int LQ = DH + 8;   // 136 halfs
  constexpr int LP = 32 + 8;   // 40 halfs
  __shared__ alignas(16) _Float16 Qs[64][LQ];
  __shared__ alignas(16) _Float16 Ks[32][LQ];
  __shared__ alignas(16) _Float16 Vts[DH][LP];   // transposed: [dh][key]
  __shared__ alignas(16) _Float16 Ps[4][16][LP]; // per-wave P staging

  const int bh = blockIdx.x;            // b*H + h
  const int q0 = blockIdx.y * 64;
  const int tid = threadIdx.x;
  const int lane = tid & 31;
  const int wave = tid >> 5;
  const int lm = lane & 15;
  const int hi = lane >> 4;

  const _Float16* Qg = Qh + ((size_t)bh * SEQ + q0) * DH;
  const _Float16* Kg = Kh + (size_t)bh * SEQ * DH;
  const _Float16* Vg = Vh + (size_t)bh * SEQ * DH;

  // Stage Q tile (64 x 128) via async-to-LDS
  for (int c = tid; c < 1024; c += 128) {
    int row = c >> 4;
    int ko = (c & 15) << 3;
    async_ld16(Qg + (size_t)row * DH + ko, &Qs[row][ko]);
  }
  wait_async0();
  __syncthreads();

  // Q A-fragments, resident for the whole block
  v16h qf[4];
#pragma unroll
  for (int dk = 0; dk < 4; ++dk) {
    int row = wave * 16 + lm;
    v8h lo = *(const v8h*)&Qs[row][dk * 32 + hi * 8];
    v8h h8 = *(const v8h*)&Qs[row][dk * 32 + 16 + hi * 8];
    qf[dk] = cat16(lo, h8);
  }

  v8f of[8] = {};
  float mrow[8], lrow[8];
#pragma unroll
  for (int r = 0; r < 8; ++r) { mrow[r] = -3.0e38f; lrow[r] = 0.0f; }

  const int nkb = (q0 >> 5) + 2;  // key blocks: keys 0 .. q0+63
  const int qWaveMax = q0 + wave * 16 + 15;

  for (int kb = 0; kb < nkb; ++kb) {
    const int k0 = kb * 32;
    __syncthreads();   // previous block's K/V reads complete
    // Stage K (async 32x128) and V transposed (128x32, through VGPRs)
    for (int c = tid; c < 512; c += 128) {
      int row = c >> 4;
      int ko = (c & 15) << 3;
      async_ld16(Kg + (size_t)(k0 + row) * DH + ko, &Ks[row][ko]);
      v8h vv = *(const v8h*)(Vg + (size_t)(k0 + row) * DH + ko);
#pragma unroll
      for (int j = 0; j < 8; ++j) Vts[ko + j][row] = vv[j];
    }
    wait_async0();
    __syncthreads();

    if (k0 <= qWaveMax) {  // wave-uniform: skip fully-masked tiles
      // S = Q * K^T  (two 16-key sub-tiles, 4 dh chunks each)
      v8f sf[2];
#pragma unroll
      for (int kt = 0; kt < 2; ++kt) {
        v8f s = {};
#pragma unroll
        for (int dk = 0; dk < 4; ++dk) {
          int r = kt * 16 + lm;
          v8h lo = *(const v8h*)&Ks[r][dk * 32 + hi * 8];
          v8h h8 = *(const v8h*)&Ks[r][dk * 32 + 16 + hi * 8];
          s = wmma16(qf[dk], cat16(lo, h8), s);
        }
        sf[kt] = s;
      }

      // causal mask + online softmax (exp2 domain)
      const int qrowBase = q0 + wave * 16 + 8 * hi;
#pragma unroll
      for (int r = 0; r < 8; ++r) {
        int qi = qrowBase + r;
#pragma unroll
        for (int kt = 0; kt < 2; ++kt) {
          int ki = k0 + kt * 16 + lm;
          float v = sf[kt][r];
          if (ki > qi) v = -3.0e38f;
          sf[kt][r] = v;
        }
        float mx = fmaxf(sf[0][r], sf[1][r]);
#pragma unroll
        for (int off = 8; off; off >>= 1)
          mx = fmaxf(mx, __shfl_xor(mx, off, 16));
        float mnew = fmaxf(mrow[r], mx);
        float alpha = exp2f(mrow[r] - mnew);
        mrow[r] = mnew;
        float psum = 0.0f;
#pragma unroll
        for (int kt = 0; kt < 2; ++kt) {
          float p = exp2f(sf[kt][r] - mnew);
          sf[kt][r] = p;
          psum += p;
        }
#pragma unroll
        for (int off = 8; off; off >>= 1) psum += __shfl_xor(psum, off, 16);
        lrow[r] = lrow[r] * alpha + psum;
#pragma unroll
        for (int nt = 0; nt < 8; ++nt) of[nt][r] *= alpha;
      }

      // Transpose P through per-wave LDS: C-layout -> A-layout
#pragma unroll
      for (int kt = 0; kt < 2; ++kt)
#pragma unroll
        for (int r = 0; r < 8; ++r)
          Ps[wave][8 * hi + r][kt * 16 + lm] = (_Float16)sf[kt][r];
      asm volatile("s_wait_dscnt 0" ::: "memory");
      v8h plo = *(const v8h*)&Ps[wave][lm][hi * 8];
      v8h phi = *(const v8h*)&Ps[wave][lm][16 + hi * 8];
      v16h pf = cat16(plo, phi);

      // O += P(16x32) * V(32x16) per 16-wide dh tile
#pragma unroll
      for (int nt = 0; nt < 8; ++nt) {
        v8h vlo = *(const v8h*)&Vts[nt * 16 + lm][hi * 8];
        v8h vhi = *(const v8h*)&Vts[nt * 16 + lm][16 + hi * 8];
        of[nt] = wmma16(pf, cat16(vlo, vhi), of[nt]);
      }
    }
  }

  // Normalize and store into [B,S,D] f16 (column = h*128 + dh)
  const int b = bh >> 4;
  const int h = bh & (NHEADS - 1);
#pragma unroll
  for (int nt = 0; nt < 8; ++nt) {
#pragma unroll
    for (int r = 0; r < 8; ++r) {
      int q = q0 + wave * 16 + 8 * hi + r;
      int dh = nt * 16 + lm;
      float v = of[nt][r] / lrow[r];
      Oh[((size_t)b * SEQ + q) * DM + h * DH + dh] = (_Float16)v;
    }
  }
}

// ---------------------------------------------------------------------------
// Launcher
// ---------------------------------------------------------------------------
extern "C" void kernel_launch(void* const* d_in, const int* in_sizes, int n_in,
                              void* d_out, int out_size, void* d_ws,
                              size_t ws_size, hipStream_t stream) {
  const float* x = (const float*)d_in[0];      // [B,S,D]
  const float* Wqkv = (const float*)d_in[1];   // [3D,D]
  const float* Wout = (const float*)d_in[2];   // [D,D]
  // d_in[3] = causal mask, computed analytically instead
  float* out = (float*)d_out;                  // [B,S,D]

  const size_t nX = (size_t)BATCH * SEQ * DM;      // 8M
  const size_t nWq = (size_t)3 * DM * DM;          // 12M
  const size_t nWo = (size_t)DM * DM;              // 4M
  const size_t nH = (size_t)BATCH * NHEADS * SEQ * DH;  // 8M

  _Float16* xh = (_Float16*)d_ws;
  _Float16* wqkvh = xh + nX;
  _Float16* wouth = wqkvh + nWq;
  _Float16* qh = wouth + nWo;
  _Float16* kh = qh + nH;
  _Float16* vh = kh + nH;
  _Float16* oh = vh + nH;

  // 1) fp32 -> fp16
  cvt_f32_f16<<<(int)(nX / 4 / 256), 256, 0, stream>>>(x, xh, (int)(nX / 4));
  cvt_f32_f16<<<(int)(nWq / 4 / 256), 256, 0, stream>>>(Wqkv, wqkvh,
                                                        (int)(nWq / 4));
  cvt_f32_f16<<<(int)(nWo / 4 / 256), 256, 0, stream>>>(Wout, wouth,
                                                        (int)(nWo / 4));

  // 2) QKV projection: [4096,2048] x [6144,2048]^T -> scatter Q/K/V
  gemm_wmma<1><<<dim3(6144 / 128, 4096 / 128), 256, 0, stream>>>(
      xh, wqkvh, nullptr, qh, kh, vh, BATCH * SEQ, 3 * DM, DM);

  // 3) causal flash attention
  flash_attn_wmma<<<dim3(BATCH * NHEADS, SEQ / 64), 128, 0, stream>>>(qh, kh,
                                                                      vh, oh);

  // 4) output projection: [4096,2048] x [2048,2048]^T -> fp32 out
  gemm_wmma<0><<<dim3(2048 / 128, 4096 / 128), 256, 0, stream>>>(
      oh, wouth, out, nullptr, nullptr, nullptr, BATCH * SEQ, DM, DM);
}
